// BigBirdForSeq_65841848647830
// MI455X (gfx1250) — compile-verified, matrix operand order checked by reading, not run on hardware
//
#include <hip/hip_runtime.h>

// ---------------- model constants ----------------
#define B_    2
#define S_    1024
#define V_    50358
#define H_    12
#define D_    64
#define HD_   768
#define FF_   3072
#define NL_   4
#define BLK_  64
#define NB_   16
#define NB2_  14
#define NR_   3
#define O_    20
#define LOPT_ 8
#define M_TOK (B_ * S_)

// ---------------- workspace layout (bytes) ----------------
#define WS_X    ((size_t)0)          // f32  [2048,768]
#define WS_XB   ((size_t)6291456)    // f16  [2048,768]
#define WS_WB   ((size_t)9437184)    // f16  weight convert buffer (max 768*3072)
#define WS_QH   ((size_t)14155776)   // f16  [2048,768]
#define WS_KH   ((size_t)17301504)   // f16  [2048,768]
#define WS_VH   ((size_t)20447232)   // f16  [2048,768]
#define WS_CTX  ((size_t)23592960)   // f16  [2048,768]
#define WS_Y    ((size_t)26738688)   // f32  [2048,768]
#define WS_HH   ((size_t)33030144)   // f16  [2048,3072]
#define WS_KIDX ((size_t)45613056)   // int  [H][NB][16]
#define WS_GLEN ((size_t)45625344)   // int  [NB]

// ---------------- WMMA fragment types ----------------
typedef _Float16 half16 __attribute__((ext_vector_type(16)));
typedef float    float8 __attribute__((ext_vector_type(8)));

union FragH { half16 v; unsigned int u[8]; _Float16 h[16]; };
union FragF { float8 v; float f[8]; };

// K-offset inside a 16x32 f16 A/B fragment for packed element pair v (0..7):
// element e -> K = (e/8)*16 + (lane>=16)*8 + (e%8); pairs (2v,2v+1) contiguous.
__device__ __forceinline__ int kloc(int v, int hi) {
  return ((v >> 2) << 4) + (hi << 3) + ((v & 3) << 1);
}

__device__ __forceinline__ float8 wmma16(half16 a, half16 b, float8 c) {
  return __builtin_amdgcn_wmma_f32_16x16x32_f16(false, a, false, b, (short)0, c, false, false);
}

// branch-free tanh: exact limits at +/-inf, no EXEC divergence in epilogue
__device__ __forceinline__ float tanh_fast(float x) {
  float e = __expf(2.0f * x);
  return 1.0f - 2.0f / (e + 1.0f);
}

__device__ __forceinline__ float gelu_tanh(float x) {
  float x3 = x * x * x;
  return 0.5f * x * (1.0f + tanh_fast(0.7978845608028654f * (x + 0.044715f * x3)));
}

__device__ __forceinline__ float block_sum_256(float v, float* red, int tid) {
  red[tid] = v; __syncthreads();
  #pragma unroll
  for (int s = 128; s > 0; s >>= 1) {
    if (tid < s) red[tid] += red[tid + s];
    __syncthreads();
  }
  float r = red[0]; __syncthreads();
  return r;
}

// ---------------- fp32 -> fp16 convert ----------------
__global__ void __launch_bounds__(256) cvt_kernel(const float* __restrict__ src,
                                                  _Float16* __restrict__ dst, int n) {
  int i = blockIdx.x * blockDim.x + threadIdx.x;
  int stride = gridDim.x * blockDim.x;
  for (; i < n; i += stride) dst[i] = (_Float16)src[i];
}

// ---------------- BigBird key-block index table ----------------
__global__ void __launch_bounds__(256) build_keyidx_kernel(const int* __restrict__ rand_idx,
                                                           int* __restrict__ kidx,
                                                           int* __restrict__ glen) {
  int t = threadIdx.x;
  if (t >= H_ * NB_) return;
  int h = t / NB_, qb = t % NB_;
  int* row = kidx + (h * NB_ + qb) * 16;
  if (qb == 0 || qb == NB_ - 1) {
    #pragma unroll
    for (int i = 0; i < 16; ++i) row[i] = i;
    if (h == 0) glen[qb] = 16;
  } else {
    row[0] = 0; row[1] = qb - 1; row[2] = qb; row[3] = qb + 1; row[4] = NB_ - 1;
    #pragma unroll
    for (int r = 0; r < NR_; ++r) {
      int rv = rand_idx[(h * NB2_ + (qb - 1)) * NR_ + r] % NB_;
      if (rv < 0) rv += NB_;
      row[5 + r] = rv;
    }
    #pragma unroll
    for (int i = 8; i < 16; ++i) row[i] = 0;
    if (h == 0) glen[qb] = 8;
  }
}

// ---------------- embedding + LayerNorm ----------------
__global__ void __launch_bounds__(256) embed_ln_kernel(
    const float* __restrict__ wemb, const float* __restrict__ pemb,
    const float* __restrict__ temb, const float* __restrict__ gg,
    const float* __restrict__ bb, const int* __restrict__ ids,
    const int* __restrict__ tts, float* __restrict__ xout,
    _Float16* __restrict__ xbout) {
  __shared__ float red[256];
  int t = blockIdx.x, tid = threadIdx.x;
  int pos = t % S_;
  size_t wo = (size_t)ids[t] * HD_;
  size_t po = (size_t)pos * HD_;
  size_t to = (size_t)tts[t] * HD_;
  float v[3];
  #pragma unroll
  for (int i = 0; i < 3; ++i) {
    int idx = tid * 3 + i;
    v[i] = wemb[wo + idx] + pemb[po + idx] + temb[to + idx];
  }
  float mean = block_sum_256(v[0] + v[1] + v[2], red, tid) * (1.0f / HD_);
  float d0 = v[0] - mean, d1 = v[1] - mean, d2 = v[2] - mean;
  float var = block_sum_256(d0 * d0 + d1 * d1 + d2 * d2, red, tid) * (1.0f / HD_);
  float rs = rsqrtf(var + 1e-12f);
  size_t base = (size_t)t * HD_;
  #pragma unroll
  for (int i = 0; i < 3; ++i) {
    int idx = tid * 3 + i;
    float o = (v[i] - mean) * rs * gg[idx] + bb[idx];
    xout[base + idx] = o;
    xbout[base + idx] = (_Float16)o;
  }
}

// ---------------- residual add + LayerNorm (in-place safe) ----------------
__global__ void __launch_bounds__(256) add_ln_kernel(
    const float* __restrict__ res, const float* __restrict__ y,
    const float* __restrict__ gg, const float* __restrict__ bb,
    float* __restrict__ xout, _Float16* __restrict__ xbout) {
  __shared__ float red[256];
  int t = blockIdx.x, tid = threadIdx.x;
  size_t base = (size_t)t * HD_;
  float v[3];
  #pragma unroll
  for (int i = 0; i < 3; ++i) {
    int idx = tid * 3 + i;
    float r = res ? res[base + idx] : 0.0f;
    v[i] = r + y[base + idx];
  }
  float mean = block_sum_256(v[0] + v[1] + v[2], red, tid) * (1.0f / HD_);
  float d0 = v[0] - mean, d1 = v[1] - mean, d2 = v[2] - mean;
  float var = block_sum_256(d0 * d0 + d1 * d1 + d2 * d2, red, tid) * (1.0f / HD_);
  float rs = rsqrtf(var + 1e-12f);
  #pragma unroll
  for (int i = 0; i < 3; ++i) {
    int idx = tid * 3 + i;
    float o = (v[i] - mean) * rs * gg[idx] + bb[idx];
    xout[base + idx] = o;
    if (xbout) xbout[base + idx] = (_Float16)o;
  }
}

// ---------------- WMMA GEMM: out = act(A[M,K] @ W[K,N] + bias) ----------------
// Block: 256 thr (8 waves); block tile 128x64; each wave owns a 32x32 region
// (2x2 WMMA tiles -> 4 v_wmma per K-step, each fragment reused twice).
// LDS double-buffered: global->LDS stores of tile k+1 overlap WMMAs on tile k.
__global__ void __launch_bounds__(256) wmma_gemm_kernel(
    const _Float16* __restrict__ A, const _Float16* __restrict__ W,
    const float* __restrict__ bias, float* __restrict__ out_f,
    _Float16* __restrict__ out_h, int M, int N, int K, int act) {
  __shared__ _Float16 As[2][128][40];  // [buf][m][k], padded
  __shared__ _Float16 Bs[2][64][40];   // [buf][n][k] (transposed), padded
  int tid = threadIdx.x;
  int wave = tid >> 5, lane = tid & 31;
  int lm = lane & 15, hi = lane >> 4;
  int m0 = blockIdx.y * 128, n0 = blockIdx.x * 64;
  int mw = (wave & 3) * 32;   // wave's row base within block tile
  int nw = (wave >> 2) * 32;  // wave's col base within block tile

  FragF c[2][2];
  #pragma unroll
  for (int im = 0; im < 2; ++im)
    #pragma unroll
    for (int in2 = 0; in2 < 2; ++in2)
      #pragma unroll
      for (int r = 0; r < 8; ++r) c[im][in2].f[r] = 0.0f;

  auto load_tiles = [&](int k0, int buf) {
    // A tile 128x32: 8 contiguous halfs per thread per pass, coalesced
    #pragma unroll
    for (int it = 0; it < 2; ++it) {
      int row = (tid >> 2) + it * 64;
      int kseg = (tid & 3) * 8;
      const unsigned int* s32 =
          (const unsigned int*)(A + (size_t)(m0 + row) * K + k0 + kseg);
      unsigned int* d32 = (unsigned int*)&As[buf][row][kseg];
      d32[0] = s32[0]; d32[1] = s32[1]; d32[2] = s32[2]; d32[3] = s32[3];
    }
    // W tile 32x64 transposed into Bs[n][k] so K-pairs are dword-contiguous
    int kk = tid >> 3, nseg = (tid & 7) * 8;
    const _Float16* src = W + (size_t)(k0 + kk) * N + n0 + nseg;
    #pragma unroll
    for (int i = 0; i < 8; ++i) Bs[buf][nseg + i][kk] = src[i];
  };

  load_tiles(0, 0);
  int buf = 0;
  for (int k0 = 0; k0 < K; k0 += 32) {
    __syncthreads();
    if (k0 + 32 < K) load_tiles(k0 + 32, buf ^ 1);  // overlap with WMMAs below
    if (k0 + 64 < K) {                              // L2 prefetch two tiles ahead
      if (tid < 128) __builtin_prefetch(A + (size_t)(m0 + tid) * K + k0 + 64, 0, 1);
      if (tid < 32)  __builtin_prefetch(W + (size_t)(k0 + 64 + tid) * N + n0, 0, 1);
    }
    FragH a0, a1, b0, b1;
    #pragma unroll
    for (int v = 0; v < 8; ++v) {
      int kl = kloc(v, hi);
      a0.u[v] = *(const unsigned int*)&As[buf][mw + lm][kl];
      a1.u[v] = *(const unsigned int*)&As[buf][mw + 16 + lm][kl];
      b0.u[v] = *(const unsigned int*)&Bs[buf][nw + lm][kl];
      b1.u[v] = *(const unsigned int*)&Bs[buf][nw + 16 + lm][kl];
    }
    c[0][0].v = wmma16(a0.v, b0.v, c[0][0].v);
    c[0][1].v = wmma16(a0.v, b1.v, c[0][1].v);
    c[1][0].v = wmma16(a1.v, b0.v, c[1][0].v);
    c[1][1].v = wmma16(a1.v, b1.v, c[1][1].v);
    buf ^= 1;
  }

  // epilogue: bias + activation; C layout: VGPR r -> M = r + hi*8, lane%16 -> N
  #pragma unroll
  for (int im = 0; im < 2; ++im) {
    #pragma unroll
    for (int in2 = 0; in2 < 2; ++in2) {
      #pragma unroll
      for (int r = 0; r < 8; ++r) {
        int m = m0 + mw + im * 16 + hi * 8 + r;
        int n = n0 + nw + in2 * 16 + lm;
        float val = c[im][in2].f[r] + bias[n];
        if (act == 1) val = gelu_tanh(val);
        size_t oidx = (size_t)m * N + n;
        if (out_f) out_f[oidx] = val;
        if (out_h) out_h[oidx] = (_Float16)val;
      }
    }
  }
}

// ---------------- BigBird block-sparse attention (flash-style) ----------------
// One workgroup per (b, h, q-block). Key list of G blocks (8 middle / 16 edge),
// streamed in 256-key chunks with online softmax; QK^T and P@V via WMMA.
__global__ void __launch_bounds__(256) attn_kernel(
    const _Float16* __restrict__ Qh, const _Float16* __restrict__ Kh,
    const _Float16* __restrict__ Vh, const float* __restrict__ mask,
    const int* __restrict__ kidx, const int* __restrict__ glen,
    _Float16* __restrict__ ctxh) {
  __shared__ _Float16 Ps[64][264];     // raw scores -> probabilities (f16)
  __shared__ float maskb[256];         // additive mask bias per chunk column
  __shared__ float mrow[64], lrow[64], arow[64];
  __shared__ float red4[4][64];
  __shared__ int klc[16];

  int qb = blockIdx.x, h = blockIdx.y, b = blockIdx.z;
  int tid = threadIdx.x;
  int wave = tid >> 5, lane = tid & 31;
  int lm = lane & 15, hi = lane >> 4;
  int mi = wave & 3, di0 = (wave >> 2) * 2;
  size_t bbase = (size_t)b * S_;
  const float scale = 0.125f;  // 1/sqrt(64)

  if (tid < 16) klc[tid] = kidx[(h * NB_ + qb) * 16 + tid];
  if (tid < 64) { mrow[tid] = -3.0e38f; lrow[tid] = 0.0f; }
  int nchunk = glen[qb] >> 2;  // 4 key-blocks (256 keys) per chunk
  __syncthreads();

  FragF acc[2];
  #pragma unroll
  for (int t2 = 0; t2 < 2; ++t2)
    #pragma unroll
    for (int r = 0; r < 8; ++r) acc[t2].f[r] = 0.0f;

  for (int ch = 0; ch < nchunk; ++ch) {
    // additive mask bias per chunk column
    {
      int kb = klc[ch * 4 + (tid >> 6)];
      int pos = kb * BLK_ + (tid & 63);
      maskb[tid] = (mask[bbase + pos] - 1.0f) * 1.0e9f;
    }
    __syncthreads();

    // ---- scores: wave computes 8 tiles (rows mi*16.., cols (wave/4)*128..) ----
    for (int j = 0; j < 8; ++j) {
      int nt = (wave >> 2) * 8 + j;
      FragF s;
      #pragma unroll
      for (int r = 0; r < 8; ++r) s.f[r] = 0.0f;
      int qrow = qb * BLK_ + mi * 16 + lm;
      int kkcol = nt * 16 + lm;
      int kb2 = klc[ch * 4 + (kkcol >> 6)];
      int kpos = kb2 * BLK_ + (kkcol & 63);
      const _Float16* qp = Qh + (bbase + qrow) * HD_ + h * D_;
      const _Float16* kp = Kh + (bbase + kpos) * HD_ + h * D_;
      #pragma unroll
      for (int ks = 0; ks < 2; ++ks) {
        FragH af, bf;
        #pragma unroll
        for (int v = 0; v < 8; ++v) {
          int kl = ks * 32 + kloc(v, hi);
          af.u[v] = *(const unsigned int*)(qp + kl);
          bf.u[v] = *(const unsigned int*)(kp + kl);
        }
        s.v = wmma16(af.v, bf.v, s.v);
      }
      #pragma unroll
      for (int r = 0; r < 8; ++r) {
        int row = mi * 16 + hi * 8 + r;
        int col = nt * 16 + lm;
        Ps[row][col] = (_Float16)(s.f[r] * scale + maskb[col]);
      }
    }
    __syncthreads();

    // ---- online softmax over 256 chunk columns ----
    {
      int row = tid & 63, c0 = (tid >> 6) * 64;
      float mx = -3.0e38f;
      for (int c = 0; c < 64; ++c) mx = fmaxf(mx, (float)Ps[row][c0 + c]);
      red4[tid >> 6][row] = mx;
    }
    __syncthreads();
    if (tid < 64) {
      float mc = fmaxf(fmaxf(red4[0][tid], red4[1][tid]),
                       fmaxf(red4[2][tid], red4[3][tid]));
      float mo = mrow[tid];
      float mn = fmaxf(mo, mc);
      float al = __expf(mo - mn);
      mrow[tid] = mn; arow[tid] = al; lrow[tid] *= al;
    }
    __syncthreads();
    {
      int row = tid & 63, c0 = (tid >> 6) * 64;
      float mn = mrow[row], sum = 0.0f;
      for (int c = 0; c < 64; ++c) {
        float p = __expf((float)Ps[row][c0 + c] - mn);
        Ps[row][c0 + c] = (_Float16)p;
        sum += p;
      }
      red4[tid >> 6][row] = sum;
    }
    __syncthreads();
    if (tid < 64)
      lrow[tid] += red4[0][tid] + red4[1][tid] + red4[2][tid] + red4[3][tid];
    __syncthreads();

    // ---- rescale running ctx accumulators, then P @ V ----
    #pragma unroll
    for (int r = 0; r < 8; ++r) {
      float al = arow[mi * 16 + hi * 8 + r];
      acc[0].f[r] *= al;
      acc[1].f[r] *= al;
    }
    for (int ks = 0; ks < 8; ++ks) {
      FragH af;
      const _Float16* pp = &Ps[mi * 16 + lm][ks * 32];
      #pragma unroll
      for (int v = 0; v < 8; ++v)
        af.u[v] = *(const unsigned int*)(pp + kloc(v, hi));
      #pragma unroll
      for (int t2 = 0; t2 < 2; ++t2) {
        int dcol = (di0 + t2) * 16 + lm;
        FragH bf;
        #pragma unroll
        for (int v = 0; v < 8; ++v) {
          int k0l = ks * 32 + kloc(v, hi);
          int k1l = k0l + 1;
          int p0 = klc[ch * 4 + (k0l >> 6)] * BLK_ + (k0l & 63);
          int p1 = klc[ch * 4 + (k1l >> 6)] * BLK_ + (k1l & 63);
          bf.h[2 * v]     = Vh[(bbase + p0) * HD_ + h * D_ + dcol];
          bf.h[2 * v + 1] = Vh[(bbase + p1) * HD_ + h * D_ + dcol];
        }
        acc[t2].v = wmma16(af.v, bf.v, acc[t2].v);
      }
    }
    __syncthreads();
  }

  // ---- epilogue: normalize and write context (f16) ----
  #pragma unroll
  for (int t2 = 0; t2 < 2; ++t2) {
    #pragma unroll
    for (int r = 0; r < 8; ++r) {
      int row = mi * 16 + hi * 8 + r;
      float l = fmaxf(lrow[row], 1e-20f);
      int qpos = qb * BLK_ + row;
      int col = h * D_ + (di0 + t2) * 16 + lm;
      ctxh[(bbase + qpos) * HD_ + col] = (_Float16)(acc[t2].f[r] / l);
    }
  }
}

// ---------------- final answer gather: 1280 dot products ----------------
__global__ void __launch_bounds__(256) answer_kernel(
    const float* __restrict__ hln, const float* __restrict__ dec_w,
    const float* __restrict__ dec_b, const int* __restrict__ opt_in,
    const float* __restrict__ opt_mask, const int* __restrict__ pos_in,
    float* __restrict__ out) {
  __shared__ float red[256];
  int bid = blockIdx.x, tid = threadIdx.x;
  int b = bid / (O_ * 4);
  int rem = bid % (O_ * 4);
  int o = rem / 4, c = rem % 4;
  int pos = pos_in[b * O_ + o];
  const float* hrow = hln + ((size_t)b * S_ + pos) * HD_;
  float accsum = 0.0f, msum = 0.0f;
  for (int j = 0; j < LOPT_; ++j) {
    size_t oidx = (size_t)((b * O_ + o) * 4 + c) * LOPT_ + j;
    int vid = opt_in[oidx];
    float mk = opt_mask[oidx];
    float part = 0.0f;
    #pragma unroll
    for (int i = 0; i < 3; ++i) {
      int idx = tid * 3 + i;
      part += hrow[idx] * dec_w[(size_t)idx * V_ + vid];
    }
    float tot = block_sum_256(part, red, tid);
    if (tid == 0) {
      accsum += (tot + dec_b[vid]) * mk;
      msum += mk;
    }
  }
  if (tid == 0) out[bid] = accsum / msum;
}

// ---------------- host-side orchestration ----------------
extern "C" void kernel_launch(void* const* d_in, const int* in_sizes, int n_in,
                              void* d_out, int out_size, void* d_ws, size_t ws_size,
                              hipStream_t stream) {
  (void)in_sizes; (void)n_in; (void)out_size; (void)ws_size;
  // input index map (setup_inputs dict insertion order; params flattened in order)
  const float* wemb  = (const float*)d_in[0];
  const float* pemb  = (const float*)d_in[1];
  const float* temb  = (const float*)d_in[2];
  const float* emb_g = (const float*)d_in[3];
  const float* emb_b = (const float*)d_in[4];
  const float* mlm_w  = (const float*)d_in[69];
  const float* mlm_b  = (const float*)d_in[70];
  const float* mlm_g  = (const float*)d_in[71];
  const float* mlm_b2 = (const float*)d_in[72];
  const float* dec_w  = (const float*)d_in[73];
  const float* dec_b  = (const float*)d_in[74];
  const int*   ids    = (const int*)d_in[75];
  const float* amask  = (const float*)d_in[76];
  const int*   tts    = (const int*)d_in[77];
  const int*   opt_in = (const int*)d_in[78];
  const float* opt_mk = (const float*)d_in[79];
  const int*   pos_in = (const int*)d_in[82];
  const int*   rand_i = (const int*)d_in[83];

  char* ws = (char*)d_ws;
  float*     x    = (float*)(ws + WS_X);
  _Float16*  xb   = (_Float16*)(ws + WS_XB);
  _Float16*  wb   = (_Float16*)(ws + WS_WB);
  _Float16*  qh   = (_Float16*)(ws + WS_QH);
  _Float16*  kh   = (_Float16*)(ws + WS_KH);
  _Float16*  vh   = (_Float16*)(ws + WS_VH);
  _Float16*  ctxh = (_Float16*)(ws + WS_CTX);
  float*     y    = (float*)(ws + WS_Y);
  _Float16*  hh   = (_Float16*)(ws + WS_HH);
  int*       kidx = (int*)(ws + WS_KIDX);
  int*       gl   = (int*)(ws + WS_GLEN);

  build_keyidx_kernel<<<1, 256, 0, stream>>>(rand_i, kidx, gl);
  embed_ln_kernel<<<M_TOK, 256, 0, stream>>>(wemb, pemb, temb, emb_g, emb_b,
                                             ids, tts, x, xb);

  dim3 g768(HD_ / 64, M_TOK / 128);
  dim3 gff(FF_ / 64, M_TOK / 128);
  dim3 gattn(NB_, H_, B_);

  for (int l = 0; l < NL_; ++l) {
    int li = 5 + l * 16;
    const float* wq  = (const float*)d_in[li + 0];
    const float* bq  = (const float*)d_in[li + 1];
    const float* wk  = (const float*)d_in[li + 2];
    const float* bk  = (const float*)d_in[li + 3];
    const float* wv  = (const float*)d_in[li + 4];
    const float* bv  = (const float*)d_in[li + 5];
    const float* wo  = (const float*)d_in[li + 6];
    const float* bo  = (const float*)d_in[li + 7];
    const float* g1  = (const float*)d_in[li + 8];
    const float* b1  = (const float*)d_in[li + 9];
    const float* w1  = (const float*)d_in[li + 10];
    const float* bf1 = (const float*)d_in[li + 11];
    const float* w2  = (const float*)d_in[li + 12];
    const float* bf2 = (const float*)d_in[li + 13];
    const float* g2  = (const float*)d_in[li + 14];
    const float* b2  = (const float*)d_in[li + 15];

    cvt_kernel<<<512, 256, 0, stream>>>(wq, wb, HD_ * HD_);
    wmma_gemm_kernel<<<g768, 256, 0, stream>>>(xb, wb, bq, nullptr, qh,
                                               M_TOK, HD_, HD_, 0);
    cvt_kernel<<<512, 256, 0, stream>>>(wk, wb, HD_ * HD_);
    wmma_gemm_kernel<<<g768, 256, 0, stream>>>(xb, wb, bk, nullptr, kh,
                                               M_TOK, HD_, HD_, 0);
    cvt_kernel<<<512, 256, 0, stream>>>(wv, wb, HD_ * HD_);
    wmma_gemm_kernel<<<g768, 256, 0, stream>>>(xb, wb, bv, nullptr, vh,
                                               M_TOK, HD_, HD_, 0);

    attn_kernel<<<gattn, 256, 0, stream>>>(qh, kh, vh, amask, kidx, gl, ctxh);

    cvt_kernel<<<512, 256, 0, stream>>>(wo, wb, HD_ * HD_);
    wmma_gemm_kernel<<<g768, 256, 0, stream>>>(ctxh, wb, bo, y, nullptr,
                                               M_TOK, HD_, HD_, 0);
    add_ln_kernel<<<M_TOK, 256, 0, stream>>>(x, y, g1, b1, x, xb);

    cvt_kernel<<<1024, 256, 0, stream>>>(w1, wb, HD_ * FF_);
    wmma_gemm_kernel<<<gff, 256, 0, stream>>>(xb, wb, bf1, nullptr, hh,
                                              M_TOK, FF_, HD_, 1);  // GELU
    cvt_kernel<<<1024, 256, 0, stream>>>(w2, wb, FF_ * HD_);
    wmma_gemm_kernel<<<g768, 256, 0, stream>>>(hh, wb, bf2, y, nullptr,
                                               M_TOK, HD_, FF_, 0);
    add_ln_kernel<<<M_TOK, 256, 0, stream>>>(x, y, g2, b2, x, xb);
  }

  // MLM head: gelu(x @ mlm_w + mlm_b) -> LN  (vocab GEMM eliminated entirely)
  cvt_kernel<<<512, 256, 0, stream>>>(mlm_w, wb, HD_ * HD_);
  wmma_gemm_kernel<<<g768, 256, 0, stream>>>(xb, wb, mlm_b, y, nullptr,
                                             M_TOK, HD_, HD_, 1);
  add_ln_kernel<<<M_TOK, 256, 0, stream>>>(nullptr, y, mlm_g, mlm_b2, x, nullptr);

  answer_kernel<<<B_ * O_ * 4, 256, 0, stream>>>(x, dec_w, dec_b, opt_in,
                                                 opt_mk, pos_in, (float*)d_out);
}
